// CurricularFace_58677843198099
// MI455X (gfx1250) — matrix operand order repaired and checked by reading, not compiled, not run on hardware
//
#include <hip/hip_runtime.h>
#include <hip/hip_bf16.h>

// ---------------------------------------------------------------------------
// CurricularFace fused pipeline for MI455X (gfx1250, wave32, WMMA bf16 split)
// out = S * epilogue( (x * rnorm) @ kernel )   with rnorm = 1/||kernel_row||
// B tiles staged LDS via Tensor Data Mover (tensor_load_to_lds + TENSORcnt).
// ---------------------------------------------------------------------------

typedef __attribute__((ext_vector_type(16))) __bf16 v16bf;
typedef __attribute__((ext_vector_type(8)))  __bf16 bf16x8;
typedef __attribute__((ext_vector_type(8)))  float  v8f;
typedef __attribute__((ext_vector_type(4)))  unsigned uint4v;
typedef __attribute__((ext_vector_type(8)))  unsigned uint8v;

#define NROW   512
#define DDIM   512
#define CCOL   100000
#define S_SCALE 64.0f
#define COS_M_C  0.87758256189037271612f
#define SIN_M_C  0.47942553860420300027f
#define THRESH_C (-0.87758256189037271612f)
#define MM_C     0.23971276930210150014f

#define TILE_LDS_BYTES (32 * 33 * 4)   // 32 rows x (32 dwords + 1 pad dword)

// ---------------- Pass A: kernel row norms (reduce over C per row d) -------
__global__ void cf_rownorm(const float* __restrict__ kern,
                           float* __restrict__ rnorm) {
  __shared__ float red[256];
  const int d = blockIdx.x;
  const float4* row = (const float4*)(kern + (size_t)d * CCOL);
  float acc = 0.f;
  for (int j = threadIdx.x; j < CCOL / 4; j += 256) {
    float4 v = row[j];
    acc += v.x * v.x + v.y * v.y + v.z * v.z + v.w * v.w;
  }
  red[threadIdx.x] = acc;
  __syncthreads();
  for (int s = 128; s > 0; s >>= 1) {
    if (threadIdx.x < s) red[threadIdx.x] += red[threadIdx.x + s];
    __syncthreads();
  }
  if (threadIdx.x == 0) rnorm[d] = rsqrtf(red[0]);
}

// ---------------- Pass B1: x' = x * rnorm, split f32 -> bf16 hi + lo -------
__global__ void cf_scale_split(const float* __restrict__ x,
                               const float* __restrict__ rnorm,
                               __bf16* __restrict__ xhi,
                               __bf16* __restrict__ xlo) {
  const int g = blockIdx.x * 256 + threadIdx.x;   // 0 .. 512*512-1
  const int d = g & (DDIM - 1);
  const float v = x[g] * rnorm[d];
  const __bf16 h = (__bf16)v;
  const __bf16 l = (__bf16)(v - (float)h);
  xhi[g] = h;
  xlo[g] = l;
}

// ---------------- Pass B2: target logits (full f32 precision) --------------
__global__ void cf_target_logit(const float* __restrict__ x,
                                const float* __restrict__ rnorm,
                                const float* __restrict__ kern,
                                const int* __restrict__ label,
                                float* __restrict__ tl) {
  __shared__ float red[256];
  const int n = blockIdx.x;
  const int lbl = label[n];
  float acc = 0.f;
  for (int d = threadIdx.x; d < DDIM; d += 256)
    acc += x[n * DDIM + d] * rnorm[d] * kern[(size_t)d * CCOL + lbl];
  red[threadIdx.x] = acc;
  __syncthreads();
  for (int s = 128; s > 0; s >>= 1) {
    if (threadIdx.x < s) red[threadIdx.x] += red[threadIdx.x + s];
    __syncthreads();
  }
  if (threadIdx.x == 0) tl[n] = fminf(1.f, fmaxf(-1.f, red[0]));
}

// ---------------- Pass B3: scalar t, per-row cos(theta+m), final logit -----
__global__ void cf_stats(const float* __restrict__ tl,
                         float* __restrict__ cosm,
                         float* __restrict__ ftl,
                         float* __restrict__ tval) {
  __shared__ float red[512];
  const int n = threadIdx.x;
  const float t0 = tl[n];
  red[n] = t0;
  __syncthreads();
  for (int s = 256; s > 0; s >>= 1) {
    if (n < s) red[n] += red[n + s];
    __syncthreads();
  }
  if (n == 0) tval[0] = 0.01f * (red[0] / (float)NROW);
  const float sn = sqrtf(fmaxf(0.f, 1.f - t0 * t0));
  const float cm = t0 * COS_M_C - sn * SIN_M_C;
  cosm[n] = cm;
  ftl[n]  = (t0 > THRESH_C) ? cm : (t0 - MM_C);
}

// ---- TDM: issue one 2D tile load (32 K-rows x 32 f32 cols) into LDS -------
// D# group0: count=1, lds_addr, 57-bit global byte addr, type=2.
// D# group1: data_size=4B, pad_enable (1 dword per 32 dwords -> 33-f32 rows),
//            tensor_dim0=C, dim0_stride=C, tensor_dim1=D, tile 32x32.
// <=2D tensor -> VADDR2/VADDR3 are NULL -> 2-operand asm form.
__device__ __forceinline__ void tdm_load_tile(const float* gptr, unsigned lds_off,
                                              uint8v g1) {
  const unsigned long long ga = (unsigned long long)(size_t)gptr;
  uint4v g0;
  g0[0] = 1u;                                      // count = 1 (valid D#)
  g0[1] = lds_off;                                 // lds_addr (bytes)
  g0[2] = (unsigned)ga;                            // global_addr[31:0]
  g0[3] = (unsigned)((ga >> 32) & 0x01FFFFFFull)   // global_addr[56:32]
        | (2u << 30);                              // type = 2 ("image")
  asm volatile("tensor_load_to_lds %0, %1" :: "s"(g0), "s"(g1) : "memory");
}

// ---------------- Pass C: WMMA bf16-split GEMM + fused epilogue ------------
// Block: 256 threads (8 waves). Output stripe: 512 rows x 32 cols.
// Wave w -> rows [w*64, w*64+64): 4 M-tiles of 16; 2 N-tiles of 16.
// K loop: 16 steps of 32. B tile TDM-staged in LDS (padded, double buffered).
__global__ void __launch_bounds__(256)
cf_gemm_epi(const __bf16* __restrict__ xhi, const __bf16* __restrict__ xlo,
            const float* __restrict__ kern, const float* __restrict__ cosm,
            const float* __restrict__ ftl,  const float* __restrict__ tval,
            const int* __restrict__ label,  float* __restrict__ out) {
  __shared__ float bt[2][32 * 33];          // 32 K-rows x 32 cols, pad to 33

  const int c0   = blockIdx.x * 32;
  const int lane = threadIdx.x & 31;
  const int wave = threadIdx.x >> 5;
  const int r0   = wave * 64;

  v8f acc[4][2] = {};

  // Scalar (SGPR) predicate: exactly one wave issues the TDM ops.
  const bool issuer = (__builtin_amdgcn_readfirstlane(threadIdx.x) < 32);

  // LDS byte offset of the tile buffers (generic LDS ptr low 32 bits).
  const unsigned lds_base = (unsigned)(size_t)(void*)&bt[0][0];

  // D# group1 (constant for all tiles)
  uint8v g1;
  g1[0] = (2u << 16)                      // data_size = 4 bytes
        | (1u << 20)                      // pad_enable
        | (4u << 22);                     // pad_interval: 32 dwords
                                          // pad_amount = 0 -> 1 dword
  g1[1] = ((unsigned)CCOL & 0xFFFFu) << 16;          // tensor_dim0[15:0]
  g1[2] = ((unsigned)CCOL >> 16) | (512u << 16);     // dim0[31:16], dim1[15:0]
  g1[3] = 32u << 16;                                 // tile_dim0 = 32
  g1[4] = 32u;                                       // tile_dim1 = 32
  g1[5] = (unsigned)CCOL;                            // dim0_stride[31:0]
  g1[6] = 0u;                                        // stride hi, dim1_stride
  g1[7] = 0u;

  if (issuer)                                        // tile 0 -> buf 0
    tdm_load_tile(kern + c0, lds_base, g1);

  const int srow = threadIdx.x >> 3;        // 0..31  (prefetch addressing)
  const int scol = (threadIdx.x & 7) * 4;

  const int m     = lane & 15;              // A: M row within 16
  const int khalf = lane >> 4;              // A: which K-half group

  for (int kk = 0; kk < 16; ++kk) {
    __builtin_amdgcn_s_wait_tensorcnt(0);   // wave0: tile kk in LDS; others: nop
    __syncthreads();                        // publish tile kk to all waves
    if (kk + 1 < 16) {
      if (issuer)                           // overlap tile kk+1 with compute kk
        tdm_load_tile(kern + (size_t)((kk + 1) * 32) * CCOL + c0,
                      lds_base + (unsigned)(((kk + 1) & 1) * TILE_LDS_BYTES), g1);
      if (kk + 2 < 16)                      // warm GL2 for the K+2 tile
        __builtin_prefetch(kern + (size_t)((kk + 2) * 32 + srow) * CCOL + c0 + scol, 0, 1);
    }

    const int k0 = kk * 32;

    // ---- A fragments for this wave's 4 M-tiles (ISA 16-bit A layout:
    // lanes 0-15 hold K {0..7,16..23}, lanes 16-31 hold K {8..15,24..31})
    v16bf Ah[4], Al[4];
    #pragma unroll
    for (int mt = 0; mt < 4; ++mt) {
      const size_t abase = (size_t)(r0 + mt * 16 + m) * DDIM + k0 + khalf * 8;
      bf16x8 h0 = *(const bf16x8*)(xhi + abase);
      bf16x8 h1 = *(const bf16x8*)(xhi + abase + 16);
      bf16x8 l0 = *(const bf16x8*)(xlo + abase);
      bf16x8 l1 = *(const bf16x8*)(xlo + abase + 16);
      Ah[mt] = __builtin_shufflevector(h0, h1, 0,1,2,3,4,5,6,7,8,9,10,11,12,13,14,15);
      Al[mt] = __builtin_shufflevector(l0, l1, 0,1,2,3,4,5,6,7,8,9,10,11,12,13,14,15);
    }

    // ---- B fragments: lane holds N column lane&15, K {0..7,16..23} / {8..15,24..31}
    const float* brow = &bt[kk & 1][khalf * 8 * 33 + (lane & 15)];
    #pragma unroll
    for (int nt = 0; nt < 2; ++nt) {
      v16bf bhi = {}, blo = {};
      #pragma unroll
      for (int j = 0; j < 16; ++j) {
        const float f = brow[((j & 7) + (j >> 3) * 16) * 33 + nt * 16];
        const __bf16 h = (__bf16)f;
        bhi[j] = h;
        blo[j] = (__bf16)(f - (float)h);
      }
      #pragma unroll
      for (int mt = 0; mt < 4; ++mt) {
        // f32-accurate split product: hi*hi + hi*lo + lo*hi
        acc[mt][nt] = __builtin_amdgcn_wmma_f32_16x16x32_bf16(
            false, Ah[mt], false, bhi, (short)0, acc[mt][nt], false, false);
        acc[mt][nt] = __builtin_amdgcn_wmma_f32_16x16x32_bf16(
            false, Ah[mt], false, blo, (short)0, acc[mt][nt], false, false);
        acc[mt][nt] = __builtin_amdgcn_wmma_f32_16x16x32_bf16(
            false, Al[mt], false, bhi, (short)0, acc[mt][nt], false, false);
      }
    }
  }

  // ---- fused CurricularFace epilogue + store -----------------------------
  // C/D layout: lanes 0-15 -> M=i, N=lane; lanes 16-31 -> M=8+i, N=lane-16
  const float t = tval[0];
  #pragma unroll
  for (int mt = 0; mt < 4; ++mt) {
    #pragma unroll
    for (int nt = 0; nt < 2; ++nt) {
      const int col = c0 + nt * 16 + (lane & 15);
      const int rb  = r0 + mt * 16 + ((lane >> 4) << 3);
      #pragma unroll
      for (int i = 0; i < 8; ++i) {
        const int r = rb + i;
        float cv = fminf(1.f, fmaxf(-1.f, acc[mt][nt][i]));
        float o  = (cv > cosm[r]) ? cv * (t + cv) : cv;
        if (label[r] == col) o = ftl[r];           // scatter override
        out[(size_t)r * CCOL + col] = o * S_SCALE;
      }
    }
  }
}

// ---------------------------------------------------------------------------
extern "C" void kernel_launch(void* const* d_in, const int* in_sizes, int n_in,
                              void* d_out, int out_size, void* d_ws, size_t ws_size,
                              hipStream_t stream) {
  const float* x     = (const float*)d_in[0];   // [512,512]
  const float* kern  = (const float*)d_in[1];   // [512,100000]
  const int*   label = (const int*)d_in[2];     // [512]
  float* out = (float*)d_out;                   // [512,100000]

  char* ws = (char*)d_ws;
  float* rnorm = (float*)(ws + 0);              // 512 f32
  float* tl    = (float*)(ws + 2048);           // 512 f32
  float* cosm  = (float*)(ws + 4096);           // 512 f32
  float* ftlv  = (float*)(ws + 6144);           // 512 f32
  float* tval  = (float*)(ws + 8192);           // 1 f32
  __bf16* xhi  = (__bf16*)(ws + 16384);                      // 512*512 bf16
  __bf16* xlo  = (__bf16*)(ws + 16384 + NROW * DDIM * 2);    // 512*512 bf16

  cf_rownorm     <<<DDIM, 256, 0, stream>>>(kern, rnorm);
  cf_scale_split <<<(NROW * DDIM) / 256, 256, 0, stream>>>(x, rnorm, xhi, xlo);
  cf_target_logit<<<NROW, 256, 0, stream>>>(x, rnorm, kern, label, tl);
  cf_stats       <<<1, 512, 0, stream>>>(tl, cosm, ftlv, tval);
  cf_gemm_epi    <<<CCOL / 32, 256, 0, stream>>>(xhi, xlo, kern, cosm, ftlv,
                                                 tval, label, out);
}
